// Vanilla_cnn_multiclassifier_49374944035207
// MI455X (gfx1250) — matrix operand order repaired and checked by reading, not compiled, not run on hardware
//
#include <hip/hip_runtime.h>
#include <hip/hip_bf16.h>
#include <hip/hip_fp16.h>

typedef __attribute__((ext_vector_type(16))) _Float16     v16h;
typedef __attribute__((ext_vector_type(8)))  float        v8f;
typedef __attribute__((ext_vector_type(8)))  unsigned int v8u;

// Build a 16xf16 WMMA fragment from two b128 register loads (pure SSA, no
// addressable temps -> no scratch).
__device__ __forceinline__ v16h make_frag(uint4 a, uint4 b) {
  v8u u = {a.x, a.y, a.z, a.w, b.x, b.y, b.z, b.w};
  return __builtin_bit_cast(v16h, u);
}

// ---------------------------------------------------------------------------
// Layer 0: direct conv 3->64, input NCHW f32 [512,3,32,32], output NHWC f16
// ---------------------------------------------------------------------------
__global__ void k_conv0(const float* __restrict__ x, const float* __restrict__ w,
                        const float* __restrict__ bias, _Float16* __restrict__ out) {
  int idx = blockIdx.x * blockDim.x + threadIdx.x;   // total = 512*32*32*64
  int oc = idx & 63;
  int xx = (idx >> 6) & 31;
  int yy = (idx >> 11) & 31;
  int b  = idx >> 16;
  float acc = bias[oc];
  #pragma unroll
  for (int r = 0; r < 3; ++r) {
    int iy = yy + r - 1;
    if (iy < 0 || iy > 31) continue;
    #pragma unroll
    for (int s = 0; s < 3; ++s) {
      int ix = xx + s - 1;
      if (ix < 0 || ix > 31) continue;
      #pragma unroll
      for (int ic = 0; ic < 3; ++ic) {
        acc += x[((b * 3 + ic) * 32 + iy) * 32 + ix] *
               w[((oc * 3 + ic) * 3 + r) * 3 + s];
      }
    }
  }
  out[idx] = (_Float16)(acc > 0.f ? acc : 0.f);
}

// ---------------------------------------------------------------------------
// Pack OIHW f32 weights into WMMA dense-B 32x16 fragment layout (f16 pairs):
// dst[((rs*KB + kb)*NT + nt)*256 + lane*8 + v]  holds K = kb*32 + (lane/16)*16
// + v*2 (+1 in high half), N = nt*16 + lane%16, for filter tap rs = r*3+s.
// ---------------------------------------------------------------------------
__global__ void k_pack(const float* __restrict__ w, unsigned int* __restrict__ dst,
                       int IC, int NT, int KB, int total) {
  int idx = blockIdx.x * blockDim.x + threadIdx.x;
  if (idx >= total) return;
  int v    = idx & 7;
  int lane = (idx >> 3) & 31;
  int rest = idx >> 8;
  int nt = rest % NT; rest /= NT;
  int kb = rest % KB;
  int rs = rest / KB;
  int r = rs / 3, s = rs % 3;
  int n    = nt * 16 + (lane & 15);
  int half = lane >> 4;
  int k0   = kb * 32 + half * 16 + v * 2;
  float f0 = w[((n * IC + k0    ) * 3 + r) * 3 + s];
  float f1 = w[((n * IC + k0 + 1) * 3 + r) * 3 + s];
  unsigned short h0 = __builtin_bit_cast(unsigned short, (_Float16)f0);
  unsigned short h1 = __builtin_bit_cast(unsigned short, (_Float16)f1);
  dst[idx] = (unsigned int)h0 | ((unsigned int)h1 << 16);
}

// ---------------------------------------------------------------------------
// Implicit-GEMM 3x3 SAME conv + bias + ReLU via V_WMMA_F32_16X16X32_F16.
// One wave computes a 16(M pixels) x 64(OC) tile: 4 N-tiles in 4 named v8f
// accumulators so each A fragment feeds 4 WMMAs.  Shapes are template
// constants (power-of-two divisions become shifts, K-loops compile-time).
// in/out: NHWC f16.  wpack: pre-packed B fragments (see k_pack).
// ---------------------------------------------------------------------------
template <int H, int W, int IC, int OC>
__global__ void k_conv_wmma(const _Float16* __restrict__ in,
                            const unsigned int* __restrict__ wpack,
                            const float* __restrict__ bias,
                            _Float16* __restrict__ out) {
  constexpr int KB = IC / 32;         // K blocks of 32
  constexpr int NT = OC / 16;         // total N tiles
  constexpr int HW = H * W;

  const int lane = threadIdx.x;       // 0..31
  const int half = lane >> 4;
  const int mloc = lane & 15;
  const int m = blockIdx.x * 16 + mloc;   // output pixel (b,y,x)
  const int b = m / HW;                    // power-of-two -> shift
  const int rem = m - b * HW;
  const int y = rem / W;
  const int x = rem - y * W;
  const int nt0 = blockIdx.y * 4;

  v8f acc0 = {};
  v8f acc1 = {};
  v8f acc2 = {};
  v8f acc3 = {};

  for (int r = 0; r < 3; ++r) {
    int iy = y + r - 1;
    for (int s = 0; s < 3; ++s) {
      int ix = x + s - 1;
      bool valid = (iy >= 0) & (iy < H) & (ix >= 0) & (ix < W);
      const _Float16* abase = in + (size_t)((b * H + iy) * W + ix) * IC;
      const int rs = r * 3 + s;
      for (int kb = 0; kb < KB; ++kb) {
        // ----- A fragment: 16x32 f16, contiguous-K lane layout (same as B), 2x b128 -----
        // VGPRs 0-3: K = kb*32 + half*16 + {0,2,4,6}(+1) -> 16B contiguous
        // VGPRs 4-7: +8
        const _Float16* p0 = abase + kb * 32 + half * 16;
        uint4 a0 = make_uint4(0u, 0u, 0u, 0u);
        uint4 a1 = make_uint4(0u, 0u, 0u, 0u);
        if (valid) {
          a0 = *(const uint4*)(p0);
          a1 = *(const uint4*)(p0 + 8);
        }
        v16h Av = make_frag(a0, a1);
        // ----- B fragments: pre-packed, 2x b128 each, 4 N-tiles -----
        const unsigned int* bb =
            wpack + (((size_t)(rs * KB + kb) * NT + nt0) << 8) + lane * 8;
        v16h B0 = make_frag(*(const uint4*)(bb +   0), *(const uint4*)(bb +   4));
        v16h B1 = make_frag(*(const uint4*)(bb + 256), *(const uint4*)(bb + 260));
        v16h B2 = make_frag(*(const uint4*)(bb + 512), *(const uint4*)(bb + 516));
        v16h B3 = make_frag(*(const uint4*)(bb + 768), *(const uint4*)(bb + 772));
        acc0 = __builtin_amdgcn_wmma_f32_16x16x32_f16(false, Av, false, B0,
                                                      (short)0, acc0, false, false);
        acc1 = __builtin_amdgcn_wmma_f32_16x16x32_f16(false, Av, false, B1,
                                                      (short)0, acc1, false, false);
        acc2 = __builtin_amdgcn_wmma_f32_16x16x32_f16(false, Av, false, B2,
                                                      (short)0, acc2, false, false);
        acc3 = __builtin_amdgcn_wmma_f32_16x16x32_f16(false, Av, false, B3,
                                                      (short)0, acc3, false, false);
      }
    }
  }

  // Epilogue: C/D layout -> VGPR j is row (j + 8*half), col = lane%16
  const int mrb = blockIdx.x * 16 + half * 8;
  auto store_tile = [&](v8f a, int t) {
    int n = (nt0 + t) * 16 + mloc;
    float bv = bias[n];
    #pragma unroll
    for (int j = 0; j < 8; ++j) {
      float val = a[j] + bv;
      out[(size_t)(mrb + j) * OC + n] = (_Float16)(val > 0.f ? val : 0.f);
    }
  };
  store_tile(acc0, 0);
  store_tile(acc1, 1);
  store_tile(acc2, 2);
  store_tile(acc3, 3);
}

// ---------------------------------------------------------------------------
// 2x2 max-pool, NHWC f16
// ---------------------------------------------------------------------------
__global__ void k_pool(const _Float16* __restrict__ in, _Float16* __restrict__ out,
                       int H, int W, int C, int total) {
  int idx = blockIdx.x * blockDim.x + threadIdx.x;
  if (idx >= total) return;
  int c = idx % C;
  int t = idx / C;
  int Wo = W >> 1, Ho = H >> 1;
  int xo = t % Wo; t /= Wo;
  int yo = t % Ho;
  int b  = t / Ho;
  size_t base = (size_t)((b * H + yo * 2) * W + xo * 2) * C + c;
  float a0 = (float)in[base];
  float a1 = (float)in[base + C];
  float a2 = (float)in[base + (size_t)W * C];
  float a3 = (float)in[base + (size_t)W * C + C];
  out[idx] = (_Float16)fmaxf(fmaxf(a0, a1), fmaxf(a2, a3));
}

// ---------------------------------------------------------------------------
// Task-indexed classifier: act NHWC f16 [512,4,4,256] -> out f32 [512,5].
// Honors torch NCHW flatten: feat[d], d = c*16 + y*4 + x.
// ---------------------------------------------------------------------------
__global__ void k_classify(const _Float16* __restrict__ act,
                           const int* __restrict__ tid,
                           const float* __restrict__ cw,
                           const float* __restrict__ cb,
                           float* __restrict__ out) {
  __shared__ float feat[4096];
  __shared__ float red[256];
  int b = blockIdx.x;
  int tx = threadIdx.x;
  int t = tid[b];
  for (int d = tx; d < 4096; d += 256) {
    int ch = d >> 4;
    int yy = (d >> 2) & 3;
    int xx = d & 3;
    feat[d] = (float)act[((b * 4 + yy) * 4 + xx) * 256 + ch];
  }
  __syncthreads();
  for (int c = 0; c < 5; ++c) {
    const float* wr = cw + (size_t)(t * 5 + c) * 4096;
    float p = 0.f;
    for (int d = tx; d < 4096; d += 256) p += feat[d] * wr[d];
    red[tx] = p;
    __syncthreads();
    for (int off = 128; off > 0; off >>= 1) {
      if (tx < off) red[tx] += red[tx + off];
      __syncthreads();
    }
    if (tx == 0) out[b * 5 + c] = red[0] + cb[t * 5 + c];
    __syncthreads();
  }
}

// ---------------------------------------------------------------------------
extern "C" void kernel_launch(void* const* d_in, const int* in_sizes, int n_in,
                              void* d_out, int out_size, void* d_ws, size_t ws_size,
                              hipStream_t stream) {
  (void)in_sizes; (void)n_in; (void)out_size; (void)ws_size;
  const float* x   = (const float*)d_in[0];
  const int*   tid = (const int*)d_in[1];
  const float* w0  = (const float*)d_in[2];
  const float* b0  = (const float*)d_in[3];
  const float* wL[5]; const float* bL[5];
  for (int i = 0; i < 5; ++i) {
    wL[i] = (const float*)d_in[4 + 2 * i];
    bL[i] = (const float*)d_in[5 + 2 * i];
  }
  const float* cw = (const float*)d_in[14];
  const float* cb = (const float*)d_in[15];
  float* out = (float*)d_out;

  char* ws = (char*)d_ws;
  const size_t BUF = (size_t)512 * 32 * 32 * 64 * 2;   // 64 MB per ping-pong buffer
  _Float16* bufA = (_Float16*)ws;
  _Float16* bufB = (_Float16*)(ws + BUF);

  static const int IC_[5] = { 64,  64, 128, 128, 256};
  static const int OC_[5] = { 64, 128, 128, 256, 256};
  unsigned int* pk[5];
  int tot_u32[5];
  size_t off = 2 * BUF;
  for (int i = 0; i < 5; ++i) {
    int KB = IC_[i] >> 5, NT = OC_[i] >> 4;
    tot_u32[i] = 9 * KB * NT * 256;
    pk[i] = (unsigned int*)(ws + off);
    off += (size_t)tot_u32[i] * 4;
  }

  // Pack conv1..conv5 weights into WMMA B-fragment layout (tiny, L2-resident).
  for (int i = 0; i < 5; ++i) {
    int KB = IC_[i] >> 5, NT = OC_[i] >> 4;
    k_pack<<<(tot_u32[i] + 255) / 256, 256, 0, stream>>>(wL[i], pk[i], IC_[i], NT, KB, tot_u32[i]);
  }

  // conv0: x (NCHW f32) -> bufA (NHWC f16 [512,32,32,64])
  k_conv0<<<(512 * 32 * 32 * 64) / 256, 256, 0, stream>>>(x, w0, b0, bufA);

  // conv1: 32x32, 64->64   (grid.y = OC/64)
  { dim3 g((512 * 32 * 32) / 16, 64 / 64);
    k_conv_wmma<32, 32, 64, 64><<<g, 32, 0, stream>>>(bufA, pk[0], bL[0], bufB); }
  // pool -> [512,16,16,64]
  { int tot = 512 * 16 * 16 * 64;
    k_pool<<<(tot + 255) / 256, 256, 0, stream>>>(bufB, bufA, 32, 32, 64, tot); }
  // conv2: 16x16, 64->128
  { dim3 g((512 * 16 * 16) / 16, 128 / 64);
    k_conv_wmma<16, 16, 64, 128><<<g, 32, 0, stream>>>(bufA, pk[1], bL[1], bufB); }
  // conv3: 16x16, 128->128
  { dim3 g((512 * 16 * 16) / 16, 128 / 64);
    k_conv_wmma<16, 16, 128, 128><<<g, 32, 0, stream>>>(bufB, pk[2], bL[2], bufA); }
  // pool -> [512,8,8,128]
  { int tot = 512 * 8 * 8 * 128;
    k_pool<<<(tot + 255) / 256, 256, 0, stream>>>(bufA, bufB, 16, 16, 128, tot); }
  // conv4: 8x8, 128->256
  { dim3 g((512 * 8 * 8) / 16, 256 / 64);
    k_conv_wmma<8, 8, 128, 256><<<g, 32, 0, stream>>>(bufB, pk[3], bL[3], bufA); }
  // conv5: 8x8, 256->256
  { dim3 g((512 * 8 * 8) / 16, 256 / 64);
    k_conv_wmma<8, 8, 256, 256><<<g, 32, 0, stream>>>(bufA, pk[4], bL[4], bufB); }
  // pool -> [512,4,4,256]
  { int tot = 512 * 4 * 4 * 256;
    k_pool<<<(tot + 255) / 256, 256, 0, stream>>>(bufB, bufA, 8, 8, 256, tot); }

  // classifier
  k_classify<<<512, 256, 0, stream>>>(bufA, tid, cw, cb, out);
}
